// BertSelfAttention_v1_23089744183835
// MI455X (gfx1250) — matrix operand order repaired and checked by reading, not compiled, not run on hardware
//
#include <hip/hip_runtime.h>

typedef __attribute__((ext_vector_type(16))) _Float16 v16h;
typedef __attribute__((ext_vector_type(8)))  _Float16 h8;
typedef __attribute__((ext_vector_type(8)))  float    v8f;

#define BB   8
#define S    512
#define HID  768
#define NH   12
#define HD   64
#define N_CTX 3145728   // 8*512*768
#define N_SC  25165824  // 8*12*512*512

// Build a 16x32 f16 A-fragment: lane holds K=[klo..klo+15] contiguous.
__device__ __forceinline__ v16h a_frag(const _Float16* p) {
  h8 lo = *(const h8*)(p);
  h8 hi = *(const h8*)(p + 8);
  return __builtin_shufflevector(lo, hi, 0,1,2,3,4,5,6,7,8,9,10,11,12,13,14,15);
}

__global__ void cvt_f32_f16(const float* __restrict__ src, _Float16* __restrict__ dst, int n) {
  int i = blockIdx.x * blockDim.x + threadIdx.x;
  if (i < n) dst[i] = (_Float16)src[i];
}

// y = x @ W^T + b (f16 in, f32 accum, f16 out). 16x64 tile per wave: 4 independent
// accumulator chains hide the WMMA->WMMA RAW hazard; A-fragment reused 4x.
// grid(HID/64=12, 256, 3), block 32 (1 wave).
__global__ void qkv_gemm(const _Float16* __restrict__ x, const _Float16* __restrict__ W16,
                         const float* __restrict__ bq, const float* __restrict__ bk,
                         const float* __restrict__ bv, _Float16* __restrict__ qkv) {
  const int lane  = threadIdx.x;
  const int obase = blockIdx.x * 64;
  const int mbase = blockIdx.y * 16;
  const int w     = blockIdx.z;

  const _Float16* Wp  = W16 + (size_t)w * (HID * HID);
  const float*   bias = (w == 0) ? bq : (w == 1) ? bk : bv;
  _Float16*      dst  = qkv + (size_t)w * N_CTX;

  const int aoff = (lane < 16) ? 0 : 16;
  const int boff = (lane < 16) ? 0 : 16;
  const _Float16* ap = x  + (size_t)(mbase + (lane & 15)) * HID + aoff;
  const _Float16* bp = Wp + (size_t)(obase + (lane & 15)) * HID + boff;

  v8f acc[4] = {};
#pragma unroll 2
  for (int kb = 0; kb < HID / 32; ++kb) {
    v16h A = a_frag(ap + kb * 32);
#pragma unroll
    for (int t = 0; t < 4; ++t) {
      v16h B = *(const v16h*)(bp + (size_t)t * 16 * HID + kb * 32);
      acc[t] = __builtin_amdgcn_wmma_f32_16x16x32_f16(false, A, false, B, (short)0, acc[t], false, false);
    }
  }
  const int m0 = mbase + ((lane < 16) ? 0 : 8);
#pragma unroll
  for (int t = 0; t < 4; ++t) {
    const int ocol = obase + t * 16 + (lane & 15);
    const float bbv = bias[ocol];
#pragma unroll
    for (int j = 0; j < 8; ++j)
      dst[(size_t)(m0 + j) * HID + ocol] = (_Float16)(acc[t][j] + bbv);
  }
}

// v [B*S, 768] -> vT [B, H, D, S]  (makes PV B-matrix columns K-contiguous)
__global__ void transpose_v(const _Float16* __restrict__ v, _Float16* __restrict__ vT) {
  int i = blockIdx.x * blockDim.x + threadIdx.x;   // over N_CTX
  int b = i / (S * HID);
  int r = i % (S * HID);
  int s = r / HID;
  int c = r % HID;
  int h = c / HD;
  int d = c % HD;
  vT[((size_t)(b * NH + h) * HD + d) * S + s] = v[i];
}

// Four score tensors: which = 0:QK 1:QQ 2:KK 3:VV.  16x64 tile per wave.
// grid(S/64=8, S/16=32, 4*96), block 32.  This kernel carries the 403MB store traffic.
__global__ void scores_kernel(const _Float16* __restrict__ qkv, const float* __restrict__ mask,
                              float* __restrict__ out_sc) {
  const int lane  = threadIdx.x;
  const int nbase = blockIdx.x * 64;
  const int mbase = blockIdx.y * 16;
  const int z     = blockIdx.z;
  const int which = z / (BB * NH);
  const int bh    = z % (BB * NH);
  const int b     = bh / NH;
  const int h     = bh % NH;

  const _Float16* q16 = qkv;
  const _Float16* k16 = qkv + (size_t)N_CTX;
  const _Float16* v16 = qkv + (size_t)2 * N_CTX;
  const _Float16* Ab = (which == 2) ? k16 : (which == 3) ? v16 : q16;
  const _Float16* Bb = (which == 0) ? k16 : (which == 1) ? q16 : (which == 2) ? k16 : v16;

  const int aoff = (lane < 16) ? 0 : 16;
  const int boff = (lane < 16) ? 0 : 16;
  const _Float16* ap = Ab + (size_t)(b * S + mbase + (lane & 15)) * HID + h * HD + aoff;
  const _Float16* bp = Bb + (size_t)(b * S + nbase + (lane & 15)) * HID + h * HD + boff;

  v8f acc[4] = {};
#pragma unroll
  for (int kb = 0; kb < 2; ++kb) {   // K = D = 64
    v16h A = a_frag(ap + kb * 32);
#pragma unroll
    for (int t = 0; t < 4; ++t) {
      v16h B = *(const v16h*)(bp + (size_t)t * 16 * HID + kb * 32);
      acc[t] = __builtin_amdgcn_wmma_f32_16x16x32_f16(false, A, false, B, (short)0, acc[t], false, false);
    }
  }
  const int m0 = mbase + ((lane < 16) ? 0 : 8);
#pragma unroll
  for (int t = 0; t < 4; ++t) {
    const int ncol = nbase + t * 16 + (lane & 15);
    const float mv = mask[b * S + ncol];
    float* dst = out_sc + (size_t)which * N_SC + ((size_t)bh * S + m0) * S + ncol;
#pragma unroll
    for (int j = 0; j < 8; ++j)
      dst[(size_t)j * S] = acc[t][j] * 0.125f + mv;   // 1/sqrt(64)
  }
}

// Fused softmax (over QK scores already in d_out) + ctx = probs @ v.  grid(32, 96), block 32.
__global__ void softmax_ctx(const float* __restrict__ scores, const _Float16* __restrict__ vT,
                            float* __restrict__ ctx) {
  __shared__ _Float16 P[16][S];   // 16 KB: normalized probs, f16
  const int lane  = threadIdx.x;
  const int mbase = blockIdx.x * 16;
  const int bh    = blockIdx.y;
  const int b     = bh / NH;
  const int h     = bh % NH;

  for (int r = 0; r < 16; ++r) {
    const float* srow = scores + ((size_t)bh * S + mbase + r) * S;
    float vals[16];
    float mx = -3.0e38f;
#pragma unroll
    for (int i = 0; i < 16; ++i) { vals[i] = srow[lane + 32 * i]; mx = fmaxf(mx, vals[i]); }
#pragma unroll
    for (int off = 16; off >= 1; off >>= 1) mx = fmaxf(mx, __shfl_xor(mx, off, 32));
    float sum = 0.f;
#pragma unroll
    for (int i = 0; i < 16; ++i) { vals[i] = __expf(vals[i] - mx); sum += vals[i]; }
#pragma unroll
    for (int off = 16; off >= 1; off >>= 1) sum += __shfl_xor(sum, off, 32);
    const float inv = 1.0f / sum;
#pragma unroll
    for (int i = 0; i < 16; ++i) P[r][lane + 32 * i] = (_Float16)(vals[i] * inv);
  }
  __syncthreads();

  const int aoff = (lane < 16) ? 0 : 16;
  const int boff = (lane < 16) ? 0 : 16;
  const int n    = lane & 15;

  // kb-outer / dblk-inner: A-fragment read once per K-block, 4 independent WMMA chains.
  v8f acc[4] = {};
#pragma unroll 2
  for (int kb = 0; kb < S / 32; ++kb) {
    v16h A = a_frag(&P[lane & 15][kb * 32 + aoff]);   // probs tile from LDS
#pragma unroll
    for (int dblk = 0; dblk < 4; ++dblk) {
      const _Float16* vcol = vT + ((size_t)bh * HD + dblk * 16 + n) * S + boff;
      v16h B = *(const v16h*)(vcol + kb * 32);        // v^T: K-contiguous
      acc[dblk] = __builtin_amdgcn_wmma_f32_16x16x32_f16(false, A, false, B, (short)0, acc[dblk], false, false);
    }
  }
#pragma unroll
  for (int dblk = 0; dblk < 4; ++dblk) {
    float* dst = ctx + ((size_t)(b * S) + mbase + ((lane < 16) ? 0 : 8)) * HID
               + h * HD + dblk * 16 + n;
#pragma unroll
    for (int j = 0; j < 8; ++j) dst[(size_t)j * HID] = acc[dblk][j];
  }
}

extern "C" void kernel_launch(void* const* d_in, const int* in_sizes, int n_in,
                              void* d_out, int out_size, void* d_ws, size_t ws_size,
                              hipStream_t stream) {
  (void)in_sizes; (void)n_in; (void)out_size; (void)ws_size;
  const float* hidden = (const float*)d_in[0];
  const float* mask   = (const float*)d_in[1];
  const float* Wq = (const float*)d_in[2];
  const float* bq = (const float*)d_in[3];
  const float* Wk = (const float*)d_in[4];
  const float* bk = (const float*)d_in[5];
  const float* Wv = (const float*)d_in[6];
  const float* bv = (const float*)d_in[7];
  float* out = (float*)d_out;

  // workspace layout (f16):
  _Float16* ws       = (_Float16*)d_ws;
  _Float16* hidden16 = ws;                                 // N_CTX
  _Float16* W16      = hidden16 + (size_t)N_CTX;           // 3 * 768*768
  _Float16* qkv16    = W16 + (size_t)3 * HID * HID;        // 3 * N_CTX  (q,k,v)
  _Float16* vT16     = qkv16 + (size_t)3 * N_CTX;          // N_CTX

  cvt_f32_f16<<<dim3((N_CTX + 255) / 256), 256, 0, stream>>>(hidden, hidden16, N_CTX);
  cvt_f32_f16<<<dim3((HID * HID + 255) / 256), 256, 0, stream>>>(Wq, W16, HID * HID);
  cvt_f32_f16<<<dim3((HID * HID + 255) / 256), 256, 0, stream>>>(Wk, W16 + (size_t)HID * HID, HID * HID);
  cvt_f32_f16<<<dim3((HID * HID + 255) / 256), 256, 0, stream>>>(Wv, W16 + (size_t)2 * HID * HID, HID * HID);

  qkv_gemm<<<dim3(HID / 64, (BB * S) / 16, 3), 32, 0, stream>>>(hidden16, W16, bq, bk, bv, qkv16);
  transpose_v<<<dim3(N_CTX / 256), 256, 0, stream>>>(qkv16 + (size_t)2 * N_CTX, vT16);
  scores_kernel<<<dim3(S / 64, S / 16, 4 * BB * NH), 32, 0, stream>>>(qkv16, mask, out + N_CTX);
  softmax_ctx<<<dim3(S / 16, BB * NH), 32, 0, stream>>>(out + N_CTX, vT16, out);
}
